// LSTM_85761906966746
// MI455X (gfx1250) — compile-verified, matrix operand order checked by reading, not run on hardware
//
#include <hip/hip_runtime.h>

#define BATCH 8192
#define SEQT  28
#define FEATN 28
#define HS    512
#define GATES 2048            // 4*HS
#define KTOT  544             // 32 (padded x) + 512 (h)
#define NKC   17              // KTOT/32
#define NCT   128             // GATES/16
#define BT    32              // batch rows per workgroup = 2 row-tiles of 16
#define NRT   (BATCH/16)      // 512 row-tiles of 16 in the whole batch
#define HSTR  520             // padded h row stride in halves (bank-conflict pad)

typedef __attribute__((ext_vector_type(16))) _Float16 v16h;
typedef __attribute__((ext_vector_type(8)))  _Float16 v8h;
typedef __attribute__((ext_vector_type(8)))  float    v8f;

union AF { v16h v; v8h h[2]; };

__device__ __forceinline__ float sigmoidf_(float x) {
    return 1.0f / (1.0f + __expf(-x));
}
__device__ __forceinline__ float tanhf_(float x) {
    float e = __expf(2.0f * x);
    return (e - 1.0f) / (e + 1.0f);
}

// ---------------------------------------------------------------------------
// Pre-pass 1: pack [W(28) ; pad(4) ; U(512)] x 2048 (fp32) into f16 WMMA-B
// fragment order: tile(kc,ct) is 32x16 = 1KB; lane's 16 halves contiguous at
// lane*32B, ordered K = run*16 + half*8 + e, lane = half*16 + n.
// ---------------------------------------------------------------------------
__global__ void pack_weights(const float* __restrict__ W,
                             const float* __restrict__ U,
                             _Float16* __restrict__ wc)
{
    int idx = blockIdx.x * 256 + threadIdx.x;
    if (idx >= KTOT * GATES) return;
    int k = idx / GATES;
    int n = idx - k * GATES;

    float v = 0.0f;
    if (k < FEATN)      v = W[(size_t)k * GATES + n];
    else if (k >= 32)   v = U[(size_t)(k - 32) * GATES + n];

    int kc = k >> 5, kr = k & 31;
    int ct = n >> 4, nr = n & 15;
    int hf  = (kr >> 3) & 1;
    int run = kr >> 4;
    int e   = kr & 7;
    int lane = hf * 16 + nr;

    size_t off = ((size_t)(kc * NCT + ct)) * 512 + (size_t)lane * 16 + run * 8 + e;
    wc[off] = (_Float16)v;
}

// ---------------------------------------------------------------------------
// Pre-pass 2: pack x [8192,28,28] fp32 into per-(row-tile, t) f16 A-fragments
// (K padded 28 -> 32). One thread per (rt, t, lane); writes 16 halves (32B).
// ---------------------------------------------------------------------------
__global__ void pack_x(const float* __restrict__ x, _Float16* __restrict__ xf)
{
    int idx = blockIdx.x * 256 + threadIdx.x;       // < 512*28*32
    if (idx >= NRT * SEQT * 32) return;
    int rt  = idx / (SEQT * 32);
    int rem = idx - rt * (SEQT * 32);
    int tt  = rem >> 5;
    int ln  = rem & 31;
    int m   = ln & 15;
    int hf  = ln >> 4;

    _Float16* dst = xf + ((size_t)(rt * SEQT + tt)) * 512 + ln * 16;
    #pragma unroll
    for (int run = 0; run < 2; ++run) {
        #pragma unroll
        for (int e = 0; e < 8; ++e) {
            int K = run * 16 + hf * 8 + e;
            float v = (K < FEATN)
                    ? x[(size_t)(rt * 16 + m) * (SEQT * FEATN) + tt * FEATN + K]
                    : 0.0f;
            dst[run * 8 + e] = (_Float16)v;
        }
    }
}

// ---------------------------------------------------------------------------
// Main LSTM kernel: 256 workgroups x 256 threads (8 waves, wave32).
// Each WG owns 32 batch rows (2 row-tiles) for all 28 timesteps.
// Wave w owns gate columns j in [w*64, w*64+64) for BOTH row-tiles:
//   acc[r][g][s]  (2 x 4 x 4 = 32 tiles), each B tile reused for r=0,1.
// Accumulators are initialized with a bias splat (WMMA C input), so the
// elementwise tail has no bias adds.
// ---------------------------------------------------------------------------
__global__ __launch_bounds__(256, 1)
void lstm_wmma(const _Float16* __restrict__ xf,
               const float* __restrict__ bias,
               const _Float16* __restrict__ wc,
               float* __restrict__ out)
{
    __shared__ __align__(32) _Float16 lds_h[BT * HSTR];   // 33.3 KB

    const int tid   = threadIdx.x;
    const int wave  = tid >> 5;     // 0..7
    const int lane  = tid & 31;
    const int lmod  = lane & 15;    // A-row / C-column index
    const int halfq = lane >> 4;
    const int b0    = blockIdx.x * BT;
    const int rt0   = blockIdx.x * 2;

    // h0 = 0
    for (int i = tid; i < BT * HSTR; i += 256) lds_h[i] = (_Float16)0.0f;

    // Bias for this wave's 16 column tiles: col = g*512 + wave*64 + s*16 + n
    float bv[4][4];
    #pragma unroll
    for (int g = 0; g < 4; ++g)
        #pragma unroll
        for (int s = 0; s < 4; ++s)
            bv[g][s] = bias[g * HS + wave * 64 + s * 16 + lmod];

    v8f zero = {};
    v8f cst[2][4];
    #pragma unroll
    for (int r = 0; r < 2; ++r)
        #pragma unroll
        for (int s = 0; s < 4; ++s)
            cst[r][s] = zero;

    __syncthreads();

    #pragma unroll 1
    for (int t = 0; t < SEQT; ++t) {
        // acc init = bias broadcast (same column bias for all 8 rows a lane holds)
        v8f acc[2][4][4];
        #pragma unroll
        for (int g = 0; g < 4; ++g)
            #pragma unroll
            for (int s = 0; s < 4; ++s) {
                v8f ab;
                #pragma unroll
                for (int e = 0; e < 8; ++e) ab[e] = bv[g][s];
                acc[0][g][s] = ab;
                acc[1][g][s] = ab;
            }

        // K loop: kc=0 is the padded x-projection chunk (A from global
        // fragment buffer), kc=1..16 are the recurrent h chunks (A from LDS).
        #pragma unroll 1
        for (int kc = 0; kc < NKC; ++kc) {
            const _Float16* wck = wc + (size_t)kc * (NCT * 512);

            // 16 B tiles into distinct registers -> one load clause,
            // overlapped with the previous chunk's WMMA drain.
            v16h b[4][4];
            #pragma unroll
            for (int g = 0; g < 4; ++g)
                #pragma unroll
                for (int s = 0; s < 4; ++s) {
                    int ct = g * 32 + wave * 4 + s;
                    b[g][s] = *(const v16h*)&wck[(size_t)ct * 512 + lane * 16];
                }

            if (kc + 1 < NKC)
                __builtin_prefetch(wc + (size_t)(kc + 1) * (NCT * 512)
                                      + (size_t)(wave * 16) * 512 + lane * 16, 0, 0);

            AF a[2];
            if (kc == 0) {
                #pragma unroll
                for (int r = 0; r < 2; ++r)
                    a[r].v = *(const v16h*)&xf[((size_t)(rt0 + r) * SEQT + t) * 512
                                               + lane * 16];
            } else {
                const int kb = (kc - 1) * 32;
                #pragma unroll
                for (int r = 0; r < 2; ++r) {
                    a[r].h[0] = *(const v8h*)&lds_h[(r * 16 + lmod) * HSTR + kb
                                                    + halfq * 8];
                    a[r].h[1] = *(const v8h*)&lds_h[(r * 16 + lmod) * HSTR + kb + 16
                                                    + halfq * 8];
                }
            }

            // 32 WMMAs; consume B tiles in load order, reuse each for r=0,1.
            #pragma unroll
            for (int g = 0; g < 4; ++g)
                #pragma unroll
                for (int s = 0; s < 4; ++s)
                    #pragma unroll
                    for (int r = 0; r < 2; ++r)
                        acc[r][g][s] = __builtin_amdgcn_wmma_f32_16x16x32_f16(
                            false, a[r].v, false, b[g][s], (short)0,
                            acc[r][g][s], false, false);
        }

        // Gate nonlinearities + state update in registers BEFORE the barrier,
        // so only the h stores sit in the barrier-to-barrier critical section.
        // C/D layout: lane n = lane&15 (column), row = halfq*8 + e.
        v8f hv[2][4];
        #pragma unroll
        for (int r = 0; r < 2; ++r) {
            #pragma unroll
            for (int s = 0; s < 4; ++s) {
                #pragma unroll
                for (int e = 0; e < 8; ++e) {
                    float ig = sigmoidf_(acc[r][0][s][e]);
                    float fg = sigmoidf_(acc[r][1][s][e]);
                    float gg = tanhf_  (acc[r][2][s][e]);
                    float og = sigmoidf_(acc[r][3][s][e]);
                    float cc = fg * cst[r][s][e] + ig * gg;
                    cst[r][s][e] = cc;
                    hv[r][s][e] = og * tanhf_(cc);
                }
            }
        }

        __syncthreads();   // all reads of lds_h complete before overwrite

        #pragma unroll
        for (int r = 0; r < 2; ++r)
            #pragma unroll
            for (int s = 0; s < 4; ++s)
                #pragma unroll
                for (int e = 0; e < 8; ++e) {
                    int row = r * 16 + halfq * 8 + e;
                    lds_h[row * HSTR + wave * 64 + s * 16 + lmod] =
                        (_Float16)hv[r][s][e];
                }

        __syncthreads();   // new h visible to all waves
    }

    // Final h -> out [8192 x 512] fp32 (coalesced over j)
    for (int i = tid; i < BT * HS; i += 256) {
        int mm = i >> 9;          // / 512
        int j  = i & (HS - 1);
        out[(size_t)(b0 + mm) * HS + j] = (float)lds_h[mm * HSTR + j];
    }
}

// ---------------------------------------------------------------------------
extern "C" void kernel_launch(void* const* d_in, const int* in_sizes, int n_in,
                              void* d_out, int out_size, void* d_ws, size_t ws_size,
                              hipStream_t stream)
{
    const float* x    = (const float*)d_in[0];   // [8192,28,28]
    const float* W    = (const float*)d_in[1];   // [28,2048]
    const float* U    = (const float*)d_in[2];   // [512,2048]
    const float* bias = (const float*)d_in[3];   // [2048]
    float* out = (float*)d_out;

    _Float16* wc = (_Float16*)d_ws;                      // 2.23 MB packed weights
    _Float16* xf = wc + (size_t)KTOT * GATES;            // 14.7 MB packed x frags

    int wtotal = KTOT * GATES;
    pack_weights<<<(wtotal + 255) / 256, 256, 0, stream>>>(W, U, wc);

    int xtotal = NRT * SEQT * 32;
    pack_x<<<(xtotal + 255) / 256, 256, 0, stream>>>(x, xf);

    lstm_wmma<<<BATCH / BT, 256, 0, stream>>>(xf, bias, wc, out);
}
